// MultiHeadAttention_19628000543310
// MI455X (gfx1250) — compile-verified, matrix operand order checked by reading, not run on hardware
//
#include <hip/hip_runtime.h>
#include <hip/hip_bf16.h>

// Shapes fixed by the reference.
#define BB 2
#define SS 2048
#define DDim 1024
#define HH 16
#define HDim 64
#define LDSROW 2052           // shorts per LDS score row: 2048 + 2-dword pad (bank skew)
constexpr float SCALE = 0.125f; // 64^-0.5

typedef __attribute__((ext_vector_type(16))) __bf16 v16bf;
typedef __attribute__((ext_vector_type(8)))  float  v8f;

union Frag {
    v16bf v;
    unsigned int u[8];
    unsigned short s[16];
};

__device__ inline unsigned short f2bf(float f) {
    unsigned int u = __float_as_uint(f);
    u += 0x7FFFu + ((u >> 16) & 1u);          // round-to-nearest-even
    return (unsigned short)(u >> 16);
}
__device__ inline float bf2f(unsigned short s) {
    return __uint_as_float((unsigned int)s << 16);
}

// 16x32 bf16 fragment from a row-major [rows x ld] bf16 matrix.
// lane&15 selects row (A) / column (B when fed the transposed matrix);
// lanes 16-31 hold the K+8 / K+24 halves per the CDNA5 WMMA VGPR layout.
__device__ inline v16bf load_frag_bf16(const unsigned short* __restrict__ base,
                                       int ld, int rc, int k0, int lane) {
    Frag f;
    const unsigned short* p =
        base + (size_t)(rc + (lane & 15)) * ld + (k0 + ((lane >> 4) << 3));
#pragma unroll
    for (int j = 0; j < 8; ++j) {
        int k = ((j >> 2) << 4) | ((j & 3) << 1); // 0,2,4,6,16,18,20,22
        f.u[j] = *(const unsigned int*)(p + k);   // dword = bf16 pair
    }
    return f.v;
}

// Same fragment sourced from LDS bf16 (probs rows).
__device__ inline v16bf load_frag_lds(const unsigned short* p0, int k0, int lane) {
    Frag f;
    const unsigned short* p = p0 + (lane & 15) * LDSROW + (k0 + ((lane >> 4) << 3));
#pragma unroll
    for (int j = 0; j < 8; ++j) {
        int k = ((j >> 2) << 4) | ((j & 3) << 1);
        f.u[j] = *(const unsigned int*)(p + k);
    }
    return f.v;
}

__device__ inline v8f wmma_bf16(v16bf a, v16bf b, v8f c) {
    return __builtin_amdgcn_wmma_f32_16x16x32_bf16(
        /*neg_a=*/false, a, /*neg_b=*/false, b,
        /*c_mod=*/(short)0, c, /*reuse_a=*/false, /*reuse_b=*/false);
}

// ---------------------------------------------------------------- converts
__global__ void __launch_bounds__(256)
cvt_kernel(const float* __restrict__ in, unsigned short* __restrict__ out, int n) {
    int i = blockIdx.x * 256 + threadIdx.x;
    if (i < n) out[i] = f2bf(in[i]);
}

// WT[n*DDim + k] = bf16(W[k*DDim + n]); 256 threads cover a 16x16 tile.
__global__ void __launch_bounds__(256)
transpose_cvt_kernel(const float* __restrict__ W, unsigned short* __restrict__ WT) {
    int n = blockIdx.x * 16 + (threadIdx.x & 15);
    int k = blockIdx.y * 16 + (threadIdx.x >> 4);
    WT[(size_t)n * DDim + k] = f2bf(W[(size_t)k * DDim + n]);
}

// ------------------------------------------------------------ generic GEMM
// C[M,N] = (A[M,K] * BT[N,K]^T + bias) * scale
// Each wave computes a 32x64 block: 2 A-fragments x 4 B-fragments -> 8 WMMA/k-step.
// out_mode: 0 = fp32 row-major, 1 = bf16 row-major, 2 = bf16 V^T scatter
__global__ void __launch_bounds__(128)
gemm_bf16_kernel(const unsigned short* __restrict__ A,
                 const unsigned short* __restrict__ BT,
                 const float* __restrict__ bias, float scale,
                 float* __restrict__ Cf, unsigned short* __restrict__ Cb,
                 int Mdim, int Ndim, int Kdim, int out_mode) {
    const int lane = threadIdx.x & 31;
    const int wave = threadIdx.x >> 5;
    const int row0 = blockIdx.x * 32;
    const int col0 = (blockIdx.y * 4 + wave) * 64;

    v8f acc[2][4] = {};
    for (int k = 0; k < Kdim; k += 32) {
        v16bf a0 = load_frag_bf16(A, Kdim, row0, k, lane);
        v16bf a1 = load_frag_bf16(A, Kdim, row0 + 16, k, lane);
#pragma unroll
        for (int t = 0; t < 4; ++t) {
            v16bf b = load_frag_bf16(BT, Kdim, col0 + t * 16, k, lane);
            acc[0][t] = wmma_bf16(a0, b, acc[0][t]);
            acc[1][t] = wmma_bf16(a1, b, acc[1][t]);
        }
    }

    const int n  = lane & 15;
    const int mb = (lane >> 4) * 8;
#pragma unroll
    for (int r = 0; r < 2; ++r) {
#pragma unroll
        for (int t = 0; t < 4; ++t) {
            int col  = col0 + t * 16 + n;
            float bv = bias[col];
#pragma unroll
            for (int i = 0; i < 8; ++i) {
                int m = row0 + r * 16 + mb + i;
                float vv = (acc[r][t][i] + bv) * scale;
                if (out_mode == 0) {
                    Cf[(size_t)m * Ndim + col] = vv;
                } else if (out_mode == 1) {
                    Cb[(size_t)m * Ndim + col] = f2bf(vv);
                } else {
                    int b2 = m / SS, s = m - b2 * SS;
                    int h = col / HDim, hd = col - h * HDim;
                    Cb[(size_t)((b2 * HH + h) * HDim + hd) * SS + s] = f2bf(vv);
                }
            }
        }
    }
}

// -------------------------------------------- fused attention core
// Per workgroup (128 thr / 4 waves): 16 query rows of one (b,h).
//   Phase 1: scores = Q.K^T + mask  -> LDS bf16 [16][2048]   (WMMA)
//   Phase 2: in-LDS row softmax (8 threads per row, shfl reductions)
//   Phase 3: normalize; single coalesced NT fp32 write of probs
//   Phase 4: out = P @ V from LDS A-fragments (WMMA), bf16 attn out
__global__ void __launch_bounds__(128)
attn_fused_kernel(const unsigned short* __restrict__ Qb,
                  const unsigned short* __restrict__ Kb,
                  const unsigned short* __restrict__ Vt,
                  const float* __restrict__ mask,
                  float* __restrict__ probs,
                  unsigned short* __restrict__ attnb) {
    extern __shared__ char smem[];
    unsigned short* srow = (unsigned short*)smem;          // [16][LDSROW] bf16
    float* stat = (float*)(smem + 16 * LDSROW * 2);        // inv[16]

    const int tid  = threadIdx.x;
    const int lane = tid & 31;
    const int wave = tid >> 5;
    const int bh   = blockIdx.y;
    const int b    = bh / HH, h = bh % HH;
    const int row0 = blockIdx.x * 16;

    const unsigned short* Q = Qb + (size_t)b * SS * DDim + h * HDim;
    const unsigned short* K = Kb + (size_t)b * SS * DDim + h * HDim;
    const float* Mk = mask + (size_t)b * SS * SS + (size_t)row0 * SS;

    const int n  = lane & 15;
    const int mb = (lane >> 4) * 8;

    // ---- Phase 1: each wave covers 512 key columns = 32 score tiles
    v16bf q0 = load_frag_bf16(Q, DDim, row0, 0, lane);
    v16bf q1 = load_frag_bf16(Q, DDim, row0, 32, lane);
    for (int ct = 0; ct < 32; ++ct) {
        int col0 = wave * 512 + ct * 16;
        v8f acc = {};
        acc = wmma_bf16(q0, load_frag_bf16(K, DDim, col0, 0, lane), acc);
        acc = wmma_bf16(q1, load_frag_bf16(K, DDim, col0, 32, lane), acc);
        int col = col0 + n;
#pragma unroll
        for (int i = 0; i < 8; ++i) {
            int m = mb + i;
            float v = acc[i] + (1.0f - Mk[(size_t)m * SS + col]) * -1e9f;
            srow[m * LDSROW + col] = f2bf(v);
        }
    }
    __syncthreads();

    // ---- Phase 2: softmax stats; 8 threads per row, 256 elems each
    {
        int r = tid >> 3, sub = tid & 7;
        unsigned short* rp = srow + r * LDSROW;
        float mx = -1e30f;
        for (int i = 0; i < 256; ++i) mx = fmaxf(mx, bf2f(rp[sub + 8 * i]));
        mx = fmaxf(mx, __shfl_xor(mx, 1, 32));
        mx = fmaxf(mx, __shfl_xor(mx, 2, 32));
        mx = fmaxf(mx, __shfl_xor(mx, 4, 32));
        float s = 0.f;
        for (int i = 0; i < 256; ++i) {
            float e = __expf(bf2f(rp[sub + 8 * i]) - mx);
            rp[sub + 8 * i] = f2bf(e);
            s += e;
        }
        s += __shfl_xor(s, 1, 32);
        s += __shfl_xor(s, 2, 32);
        s += __shfl_xor(s, 4, 32);
        if (sub == 0) stat[r] = 1.0f / s;
    }
    __syncthreads();

    // ---- Phase 3: normalize in LDS + single coalesced NT fp32 probs write
    float* Pg = probs + (size_t)bh * SS * SS + (size_t)row0 * SS;
    for (int i = 0; i < 256; ++i) {
        int idx = i * 128 + tid;                 // 0..32767 over [16][2048]
        int r = idx >> 11, c = idx & 2047;
        float v = bf2f(srow[r * LDSROW + c]) * stat[r];
        srow[r * LDSROW + c] = f2bf(v);
        __builtin_nontemporal_store(v, Pg + idx);
    }
    __syncthreads();

    // ---- Phase 4: P(16x2048) @ V(2048x64); wave w -> head-dim cols [16w,16w+16)
    const unsigned short* V = Vt + (size_t)bh * HDim * SS;
    v8f oacc = {};
    for (int k = 0; k < SS; k += 32) {
        v16bf a  = load_frag_lds(srow, k, lane);
        v16bf bf = load_frag_bf16(V, SS, wave * 16, k, lane);
        oacc = wmma_bf16(a, bf, oacc);
    }
#pragma unroll
    for (int i = 0; i < 8; ++i) {
        int s = row0 + mb + i;
        attnb[(size_t)(b * SS + s) * DDim + h * HDim + wave * 16 + n] = f2bf(oacc[i]);
    }
}

// ---------------------------------------------------------------- launch
extern "C" void kernel_launch(void* const* d_in, const int* in_sizes, int n_in,
                              void* d_out, int out_size, void* d_ws, size_t ws_size,
                              hipStream_t stream) {
    const float* X    = (const float*)d_in[0];
    const float* mask = (const float*)d_in[1];
    const float* Wq   = (const float*)d_in[2];
    const float* bq   = (const float*)d_in[3];
    const float* Wk   = (const float*)d_in[4];
    const float* bk   = (const float*)d_in[5];
    const float* Wv   = (const float*)d_in[6];
    const float* bv   = (const float*)d_in[7];
    const float* Wo   = (const float*)d_in[8];
    const float* bo   = (const float*)d_in[9];

    float* out   = (float*)d_out;
    float* probs = out + (size_t)BB * SS * DDim;

    // Workspace layout (bf16 shorts), 48 MB total.
    unsigned short* w = (unsigned short*)d_ws;
    const size_t MEG = 1024u * 1024u;
    unsigned short* Xb  = w;
    unsigned short* WqT = w + 4 * MEG;
    unsigned short* WkT = w + 5 * MEG;
    unsigned short* WvT = w + 6 * MEG;
    unsigned short* WoT = w + 7 * MEG;
    unsigned short* Qb  = w + 8 * MEG;
    unsigned short* Kb  = w + 12 * MEG;
    unsigned short* Vt  = w + 16 * MEG;
    unsigned short* Ab  = w + 20 * MEG;

    // 1. fp32 -> bf16 (X) and transposed bf16 weights.
    cvt_kernel<<<(BB * SS * DDim + 255) / 256, 256, 0, stream>>>(X, Xb, BB * SS * DDim);
    dim3 tb(256), tg(DDim / 16, DDim / 16);
    transpose_cvt_kernel<<<tg, tb, 0, stream>>>(Wq, WqT);
    transpose_cvt_kernel<<<tg, tb, 0, stream>>>(Wk, WkT);
    transpose_cvt_kernel<<<tg, tb, 0, stream>>>(Wv, WvT);
    transpose_cvt_kernel<<<tg, tb, 0, stream>>>(Wo, WoT);

    // 2. QKV projections (WMMA). Q pre-scaled, V stored transposed.
    dim3 gb(128);
    dim3 gg(BB * SS / 32, DDim / 256);
    gemm_bf16_kernel<<<gg, gb, 0, stream>>>(Xb, WqT, bq, SCALE, nullptr, Qb,
                                            BB * SS, DDim, DDim, 1);
    gemm_bf16_kernel<<<gg, gb, 0, stream>>>(Xb, WkT, bk, 1.0f, nullptr, Kb,
                                            BB * SS, DDim, DDim, 1);
    gemm_bf16_kernel<<<gg, gb, 0, stream>>>(Xb, WvT, bv, 1.0f, nullptr, Vt,
                                            BB * SS, DDim, DDim, 2);

    // 3. Fused scores + softmax + PV. ~64.2 KB dynamic LDS per workgroup
    //    (CDNA5 WGP has 320 KB; raise the HIP cap defensively — host-side,
    //    not stream-ordered, so graph-capture safe and idempotent).
    const int smem_bytes = 16 * LDSROW * 2 + 16 * 4;
    hipFuncSetAttribute((const void*)attn_fused_kernel,
                        hipFuncAttributeMaxDynamicSharedMemorySize, smem_bytes);
    dim3 ag(SS / 16, BB * HH);
    attn_fused_kernel<<<ag, gb, smem_bytes, stream>>>(Qb, Kb, Vt, mask, probs, Ab);

    // 4. Output projection -> fp32 d_out.
    gemm_bf16_kernel<<<gg, gb, 0, stream>>>(Ab, WoT, bo, 1.0f, out, nullptr,
                                            BB * SS, DDim, DDim, 0);
}
